// GatedDeltaNet_17248588661296
// MI455X (gfx1250) — matrix operand order repaired and checked by reading, not compiled, not run on hardware
//
#include <hip/hip_runtime.h>
#include <math.h>

// ---- problem constants (match reference) ----
#define BB   4
#define SS   2048
#define HIDD 512
#define NHH  16
#define DDIM 128
#define KW   4
#define CC   (NHH * DDIM)   // 2048
#define MM   (BB * SS)      // 8192

#define TCH    16           // scan chunk (timesteps staged per TDM transfer)
#define NCHUNK (SS / TCH)   // 128

typedef __bf16 bf16;
typedef __attribute__((ext_vector_type(16))) __bf16 bf16x16;
typedef __attribute__((ext_vector_type(8)))  float  f32x8;
typedef __attribute__((ext_vector_type(4)))  unsigned int u32x4;
typedef __attribute__((ext_vector_type(4)))  int i32x4v;
typedef __attribute__((ext_vector_type(8)))  int i32x8v;

union FragBF { bf16x16 v; u32x4 q[2]; };

#if __has_builtin(__builtin_amdgcn_tensor_load_to_lds)
#define HAVE_TDM 1
#else
#define HAVE_TDM 0
#endif

// ---------------- elementwise convert f32 -> bf16 ----------------
__global__ void gdn_cvt_bf16(const float* __restrict__ in, bf16* __restrict__ out, int n) {
  int i = blockIdx.x * blockDim.x + threadIdx.x;
  if (i < n) out[i] = (bf16)in[i];
}

// ---------------- WMMA GEMM:  Out[M,N] = A[M,Kd] * Bw[N,Kd]^T ----------------
// 8 waves / block, each wave computes a 32x64 output block (2x4 WMMA tiles):
// 6 fragment loads per 8 WMMAs (vs 1:1 for 32x32) -> ~25% less L2 traffic.
__global__ __launch_bounds__(256)
void gdn_wmma_gemm(const bf16* __restrict__ A,   // [M, Kd] row-major, bf16
                   const bf16* __restrict__ Bw,  // [N, Kd] row-major, bf16 (B transposed)
                   float* __restrict__ Out,      // [M, N] f32
                   int M, int N, int Kd) {
  const int lane = threadIdx.x & 31;
  const int w    = blockIdx.x * 8 + (threadIdx.x >> 5);
  const int n64  = N >> 6;
  const int wm   = (w / n64) << 5;
  const int wn   = (w % n64) << 6;
  if (wm >= M) return;
  const int r  = lane & 15;
  const int hi = lane >> 4;

  f32x8 acc[2][4] = {};
  for (int k0 = 0; k0 < Kd; k0 += 32) {
    FragBF a[2], b[4];
    // A fragment: 16x32 bf16. lanes 0-15: K 0..7 then 16..23; lanes 16-31: K 8..15 then 24..31
#pragma unroll
    for (int ti = 0; ti < 2; ++ti) {
      const bf16* p = A + (size_t)(wm + ti * 16 + r) * Kd + k0 + hi * 8;
      a[ti].q[0] = *(const u32x4*)p;
      a[ti].q[1] = *(const u32x4*)(p + 16);
    }
    // B fragment: 32x16 bf16, column n = lane&15, lanes 0-15: K 0..7 then 16..23; lanes 16-31: K 8..15 then 24..31
#pragma unroll
    for (int tj = 0; tj < 4; ++tj) {
      const bf16* p = Bw + (size_t)(wn + tj * 16 + r) * Kd + k0 + hi * 8;
      b[tj].q[0] = *(const u32x4*)p;
      b[tj].q[1] = *(const u32x4*)(p + 16);
    }
#pragma unroll
    for (int ti = 0; ti < 2; ++ti)
#pragma unroll
      for (int tj = 0; tj < 4; ++tj)
        acc[ti][tj] = __builtin_amdgcn_wmma_f32_16x16x32_bf16(
            false, a[ti].v, false, b[tj].v, (short)0, acc[ti][tj], false, false);
  }
  // C/D layout: VGPR e: lanes 0-15 -> M = e, lanes 16-31 -> M = 8+e; N = lane&15
#pragma unroll
  for (int ti = 0; ti < 2; ++ti)
#pragma unroll
    for (int tj = 0; tj < 4; ++tj)
#pragma unroll
      for (int e = 0; e < 8; ++e) {
        int row = wm + ti * 16 + hi * 8 + e;
        int col = wn + tj * 16 + r;
        Out[(size_t)row * N + col] = acc[ti][tj][e];
      }
}

// ---------------- causal depthwise conv (+ optional silu + l2norm over head dim) ----------------
template <bool ACT_NORM>
__global__ __launch_bounds__(128)
void gdn_conv(const float* __restrict__ raw,  // [B,S,C] f32
              const float* __restrict__ cw,   // [C,1,K]
              float* __restrict__ out) {      // [B,S,C] f32
  int blk = blockIdx.x;
  int h = blk % NHH;
  int s = (blk / NHH) % SS;
  int b = blk / (NHH * SS);
  int d = threadIdx.x;
  int c = h * DDIM + d;

  float acc = 0.f;
#pragma unroll
  for (int t = 0; t < KW; ++t) {
    int sp = s - (KW - 1) + t;
    if (sp >= 0) acc = fmaf(cw[c * KW + t], raw[((size_t)(b * SS + sp)) * CC + c], acc);
  }
  size_t oidx = ((size_t)(b * SS + s)) * CC + c;
  if (ACT_NORM) {
    float y = acc / (1.f + __expf(-acc));  // silu
    float ssq = y * y;
#pragma unroll
    for (int off = 16; off > 0; off >>= 1) ssq += __shfl_xor(ssq, off, 32);
    __shared__ float red[4];
    if ((threadIdx.x & 31) == 0) red[threadIdx.x >> 5] = ssq;
    __syncthreads();
    ssq = red[0] + red[1] + red[2] + red[3];
    float nrm = fmaxf(sqrtf(ssq), 1e-12f);
    out[oidx] = y / nrm;
  } else {
    out[oidx] = acc;
  }
}

// ---------------- beta = sigmoid(x @ w_beta^T) ----------------
__global__ __launch_bounds__(256)
void gdn_beta(const float* __restrict__ x,   // [M, HID]
              const float* __restrict__ wb,  // [NH, HID]
              float* __restrict__ beta) {    // [M, NH]
  int idx = blockIdx.x * blockDim.x + threadIdx.x;
  if (idx >= MM * NHH) return;
  int h = idx % NHH;
  int ms = idx / NHH;
  const float* xp = x + (size_t)ms * HIDD;
  const float* wp = wb + (size_t)h * HIDD;
  float acc = 0.f;
#pragma unroll 4
  for (int kk = 0; kk < HIDD; ++kk) acc = fmaf(xp[kk], wp[kk], acc);
  beta[idx] = 1.f / (1.f + __expf(-acc));
}

#if HAVE_TDM
// Issue one TDM 2D tile load: rows x rowlen f32 elements, row stride strideElems,
// global -> LDS at byte offset lds_off. D# bit layout per CDNA5 ISA ch.8.
__device__ __forceinline__ void tdm_load_tile(unsigned lds_off, const void* gptr,
                                              int rowlen, int rows, int strideElems) {
  unsigned long long ga = (unsigned long long)(uintptr_t)gptr;
  u32x4 g0;
  g0[0] = 1u;                                                 // count=1 (valid), user mode
  g0[1] = lds_off;                                            // lds_addr [63:32]
  g0[2] = (unsigned)ga;                                       // global_addr [95:64]
  g0[3] = (unsigned)((ga >> 32) & 0x01FFFFFFu) | (2u << 30);  // addr[56:32] | type=2
  i32x8v g1;
  g1[0] = (2 << 16);                                          // data_size=4B, no multicast
  g1[1] = (rowlen & 0xFFFF) << 16;                            // tensor_dim0 [79:48] lo
  g1[2] = ((rowlen >> 16) & 0xFFFF) | ((rows & 0xFFFF) << 16);// dim0 hi | tensor_dim1 lo
  g1[3] = ((rows >> 16) & 0xFFFF) | ((rowlen & 0xFFFF) << 16);// dim1 hi | tile_dim0 [127:112]
  g1[4] = (rows & 0xFFFF);                                    // tile_dim1 [143:128], tile_dim2=0
  g1[5] = strideElems;                                        // tensor_dim0_stride [191:160]
  g1[6] = 0;                                                  // stride hi | dim1_stride lo
  g1[7] = 0;
  i32x4v z4 = {0, 0, 0, 0};
#if __clang_major__ >= 23
  i32x8v z8 = {0, 0, 0, 0, 0, 0, 0, 0};
  __builtin_amdgcn_tensor_load_to_lds(g0, g1, z4, z4, z8, 0);
#else
  __builtin_amdgcn_tensor_load_to_lds(g0, g1, z4, z4, 0);
#endif
}
#endif

// ---------------- sequential delta scan, state row-split 4x, TDM double-buffered ----------------
// grid = B*NH*4 blocks; block = 256 threads. Each block owns a 32x128 slice of one
// (b,h) state in registers (16 f32/thread). q/k/v/beta for TCH timesteps are staged
// into LDS by the Tensor Data Mover (double buffered, TENSORcnt-pipelined).
__global__ __launch_bounds__(256)
void gdn_scan(const float* __restrict__ q,     // [B,S,C] normalized
              const float* __restrict__ k,     // [B,S,C] normalized
              const float* __restrict__ v,     // [B,S,C]
              const float* __restrict__ beta,  // [B,S,NH]
              float* __restrict__ out) {       // [B,S,C]
  const int slice = blockIdx.x & 3;   // 4 row-slices of 32
  const int bh    = blockIdx.x >> 2;  // 0..63
  const int b = bh / NHH, h = bh % NHH;
  const int tid  = threadIdx.x;
  const int iLoc = tid >> 3;              // 0..31 local row
  const int i    = slice * 32 + iLoc;     // state row 0..127
  const int jB   = (tid & 7) * 16;        // 16-col strip

  __shared__ float shK[2][TCH][DDIM];
  __shared__ float shQ[2][TCH][DDIM];
  __shared__ float shV[2][TCH][32];
  __shared__ float shBt[2][TCH];

  float Sreg[16];
#pragma unroll
  for (int jj = 0; jj < 16; ++jj) Sreg[jj] = 0.f;

  const size_t headOff = (size_t)h * DDIM;
  const size_t rowBase = (size_t)b * SS;

  auto issue = [&](int chunk, int buf) {
    size_t row0 = rowBase + (size_t)chunk * TCH;  // flat (b*S + t0)
    const float* kg = k + row0 * CC + headOff;
    const float* qg = q + row0 * CC + headOff;
    const float* vg = v + row0 * CC + headOff + slice * 32;
#if HAVE_TDM
    if (tid < 32) {  // wave 0 drives the DMA (TDM is a per-wave instruction)
      tdm_load_tile((unsigned)(uintptr_t)&shK[buf][0][0], kg, DDIM, TCH, CC);
      tdm_load_tile((unsigned)(uintptr_t)&shQ[buf][0][0], qg, DDIM, TCH, CC);
      tdm_load_tile((unsigned)(uintptr_t)&shV[buf][0][0], vg, 32, TCH, CC);
      if (tid < TCH) shBt[buf][tid] = beta[(row0 + tid) * NHH + h];
    }
#else
    for (int e = tid; e < TCH * DDIM; e += 256) {
      int tt = e >> 7, d = e & 127;
      shK[buf][tt][d] = kg[(size_t)tt * CC + d];
      shQ[buf][tt][d] = qg[(size_t)tt * CC + d];
    }
    for (int e = tid; e < TCH * 32; e += 256) {
      int tt = e >> 5, d = e & 31;
      shV[buf][tt][d] = vg[(size_t)tt * CC + d];
    }
    if (tid < TCH) shBt[buf][tid] = beta[(row0 + tid) * NHH + h];
#endif
  };

  issue(0, 0);
  for (int c = 0; c < NCHUNK; ++c) {
    const int buf = c & 1;
    const bool hasNext = (c + 1 < NCHUNK);
    if (hasNext) issue(c + 1, buf ^ 1);  // pipeline next chunk's DMA behind compute
#if HAVE_TDM
    if (tid < 32) {
      if (hasNext) __builtin_amdgcn_s_wait_tensorcnt(3);  // current chunk done, next in flight
      else         __builtin_amdgcn_s_wait_tensorcnt(0);
    }
#endif
    __syncthreads();

    const size_t row0 = rowBase + (size_t)c * TCH;
    for (int tt = 0; tt < TCH; ++tt) {
      const float bb = shBt[buf][tt];
      const float a  = bb * shK[buf][tt][i];     // beta * k_i
      const float cv = bb * shV[buf][tt][iLoc];  // beta * v_i
      float acc = 0.f;
#pragma unroll
      for (int jj = 0; jj < 16; ++jj) {
        float m = shK[buf][tt][jB + jj];
        float s = Sreg[jj];
        s = fmaf(-(a * m), s, s);   // S *= (1 - beta*k_i*k_j)
        s = fmaf(cv, m, s);         // += beta*v_i*k_j
        Sreg[jj] = s;
        acc = fmaf(s, shQ[buf][tt][jB + jj], acc);  // o_i partial
      }
      acc += __shfl_down(acc, 4, 8);
      acc += __shfl_down(acc, 2, 8);
      acc += __shfl_down(acc, 1, 8);
      if ((tid & 7) == 0) out[(row0 + tt) * CC + headOff + i] = acc;
    }
    __syncthreads();  // protect buf before its reuse two chunks later
  }
}

// ---------------- fused gate + convert: g = bf16(scan_out * silu(og)) ----------------
__global__ void gdn_gate_cvt(const float* __restrict__ so, const float* __restrict__ og,
                             bf16* __restrict__ g, int n) {
  int i = blockIdx.x * blockDim.x + threadIdx.x;
  if (i < n) {
    float o = og[i];
    g[i] = (bf16)(so[i] * (o / (1.f + __expf(-o))));
  }
}

// ---------------- host orchestration ----------------
extern "C" void kernel_launch(void* const* d_in, const int* in_sizes, int n_in,
                              void* d_out, int out_size, void* d_ws, size_t ws_size,
                              hipStream_t stream) {
  const float* x      = (const float*)d_in[0];
  const float* wq     = (const float*)d_in[1];
  const float* wk     = (const float*)d_in[2];
  const float* wv     = (const float*)d_in[3];
  const float* wo     = (const float*)d_in[4];
  const float* w_beta = (const float*)d_in[5];
  const float* w_og   = (const float*)d_in[6];
  const float* cqw    = (const float*)d_in[7];
  const float* ckw    = (const float*)d_in[8];
  const float* cvw    = (const float*)d_in[9];

  char* ws = (char*)d_ws;
  size_t off = 0;
  auto carve = [&](size_t bytes) { size_t r = off; off += (bytes + 255) & ~(size_t)255; return r; };
  const size_t SZ_XBF = (size_t)MM * HIDD * sizeof(bf16);
  const size_t SZ_WBF = (size_t)CC * HIDD * sizeof(bf16);
  const size_t SZ_BIG = (size_t)MM * CC * sizeof(float);
  const size_t SZ_BET = (size_t)MM * NHH * sizeof(float);
  const size_t SZ_GBF = (size_t)MM * CC * sizeof(bf16);

  bf16*  xbf   = (bf16*)(ws + carve(SZ_XBF));
  bf16*  wqbf  = (bf16*)(ws + carve(SZ_WBF));
  bf16*  wkbf  = (bf16*)(ws + carve(SZ_WBF));
  bf16*  wvbf  = (bf16*)(ws + carve(SZ_WBF));
  bf16*  wogbf = (bf16*)(ws + carve(SZ_WBF));
  bf16*  wobf  = (bf16*)(ws + carve(SZ_WBF));
  float* rawq  = (float*)(ws + carve(SZ_BIG));  // later reused for og projection
  float* rawk  = (float*)(ws + carve(SZ_BIG));  // later reused for scan output
  float* rawv  = (float*)(ws + carve(SZ_BIG));
  float* qn    = (float*)(ws + carve(SZ_BIG));
  float* kn    = (float*)(ws + carve(SZ_BIG));
  float* vc    = (float*)(ws + carve(SZ_BIG));
  float* betab = (float*)(ws + carve(SZ_BET));
  bf16*  gbf   = (bf16*)(ws + carve(SZ_GBF));
  (void)ws_size; (void)in_sizes; (void)n_in; (void)out_size;

  const int nX = MM * HIDD;   // 4,194,304
  const int nW = CC * HIDD;   // 1,048,576
  const int nC = MM * CC;     // 16,777,216

  // 1) precision conversion for the WMMA path
  gdn_cvt_bf16<<<(nX + 255) / 256, 256, 0, stream>>>(x, xbf, nX);
  gdn_cvt_bf16<<<(nW + 255) / 256, 256, 0, stream>>>(wq, wqbf, nW);
  gdn_cvt_bf16<<<(nW + 255) / 256, 256, 0, stream>>>(wk, wkbf, nW);
  gdn_cvt_bf16<<<(nW + 255) / 256, 256, 0, stream>>>(wv, wvbf, nW);
  gdn_cvt_bf16<<<(nW + 255) / 256, 256, 0, stream>>>(w_og, wogbf, nW);
  gdn_cvt_bf16<<<(nW + 255) / 256, 256, 0, stream>>>(wo, wobf, nW);

  // 2) projection GEMMs (M=8192, N=2048, K=512): (M/32)*(N/64)/8 = 1024 blocks
  const int projBlocks = (MM / 32) * (CC / 64) / 8;
  gdn_wmma_gemm<<<projBlocks, 256, 0, stream>>>(xbf, wqbf, rawq, MM, CC, HIDD);
  gdn_wmma_gemm<<<projBlocks, 256, 0, stream>>>(xbf, wkbf, rawk, MM, CC, HIDD);
  gdn_wmma_gemm<<<projBlocks, 256, 0, stream>>>(xbf, wvbf, rawv, MM, CC, HIDD);

  // 3) beta projection + sigmoid
  gdn_beta<<<(MM * NHH + 255) / 256, 256, 0, stream>>>(x, w_beta, betab);

  // 4) causal dwconv (+silu+l2norm for q,k)
  const int convBlocks = BB * SS * NHH;  // 131072
  gdn_conv<true ><<<convBlocks, 128, 0, stream>>>(rawq, cqw, qn);
  gdn_conv<true ><<<convBlocks, 128, 0, stream>>>(rawk, ckw, kn);
  gdn_conv<false><<<convBlocks, 128, 0, stream>>>(rawv, cvw, vc);

  // 5) og projection (reuse rawq)
  gdn_wmma_gemm<<<projBlocks, 256, 0, stream>>>(xbf, wogbf, rawq, MM, CC, HIDD);

  // 6) sequential delta scan (state row-split 4x -> 256 WGs, TDM double-buffered)
  gdn_scan<<<BB * NHH * 4, 256, 0, stream>>>(qn, kn, vc, betab, rawk);

  // 7) gate + bf16 convert
  gdn_gate_cvt<<<(nC + 255) / 256, 256, 0, stream>>>(rawk, rawq, gbf, nC);

  // 8) output GEMM (M=8192, N=512, K=2048) -> d_out f32
  const int outBlocks = (MM / 32) * (HIDD / 64) / 8;  // 256
  gdn_wmma_gemm<<<outBlocks, 256, 0, stream>>>(gbf, wobf, (float*)d_out, MM, HIDD, CC);
}
